// MultiScaleBlock_77043123356276
// MI455X (gfx1250) — matrix operand order repaired and checked by reading, not compiled, not run on hardware
//
#include <hip/hip_runtime.h>
#include <hip/hip_bf16.h>

// ---------------- problem constants ----------------
#define B_      8
#define HH_     56
#define WW_     56
#define C_      768
#define HEADS_  12
#define HD_     64
#define NTOK_   3137            // 56*56+1
#define HP_     28              // pooled spatial
#define NP_     785             // 28*28+1
#define MTOK_   (B_*NP_)        // 6280
#define PADK_   800             // NP_ padded to 16*50
#define KT_     50              // 800/16 k tiles
#define VSTR_   800             // token stride of transposed V

typedef __attribute__((ext_vector_type(16))) _Float16 v16h;
typedef __attribute__((ext_vector_type(8)))  _Float16 v8h;
typedef __attribute__((ext_vector_type(8)))  float    v8f;

__device__ __forceinline__ v8f wmma_f16(v16h a, v16h b, v8f c) {
  // D = A(16x32) x B(32x16) + C, f32 accum
  return __builtin_amdgcn_wmma_f32_16x16x32_f16(false, a, false, b, (short)0, c, false, false);
}

// ---------------- WMMA fragment loaders (CDNA5 wave32 layouts) ----------------
// All loads are UNCONDITIONAL; out-of-range rows/cols are index-clamped and the
// resulting lanes are masked downstream. Each fragment = two 16-byte loads.

// A 16x32 f16 (M x K): lanes 0-15 M=0..15 {K=0..7,16..23}; lanes 16-31 {K=8..15,24..31}
// Source row-major, row index clamped to [0, mmax).
__device__ __forceinline__ v16h frag_a(const _Float16* src, int lda, int mmax, int kbase) {
  int lane = threadIdx.x & 31;
  int m = lane & 15;
  if (m >= mmax) m = mmax - 1;
  int koff = (lane & 16) ? 8 : 0;
  const _Float16* p = src + (size_t)m * lda + kbase + koff;
  v8h lo = *(const v8h*)p;          // K = koff .. koff+7
  v8h hi = *(const v8h*)(p + 16);   // K = koff+16 .. koff+23
  v16h r;
#pragma unroll
  for (int i = 0; i < 8; ++i) { r[i] = lo[i]; r[i + 8] = hi[i]; }
  return r;
}

// A fragment from f32 LDS with per-row scale (softmax 1/sum folded into cvt)
__device__ __forceinline__ v16h frag_a_f32s(const float* src, int lda, const float* rscale) {
  int lane = threadIdx.x & 31;
  int m = lane & 15;
  int koff = (lane & 16) ? 8 : 0;
  float s = rscale[m];
  const float* p = src + (size_t)m * lda + koff;
  v16h r;
#pragma unroll
  for (int i = 0; i < 8; ++i) {
    r[i]     = (_Float16)(p[i] * s);
    r[i + 8] = (_Float16)(p[i + 16] * s);
  }
  return r;
}

// B 32x16 f16 (K x N) where memory is N-major: src[n*ldn + k] (contiguous in K).
// lanes 0-15: K={0..7,16..23}, lanes 16-31: K={8..15,24..31}, N = lane%16 (clamped to nmax).
__device__ __forceinline__ v16h frag_b_nmajor(const _Float16* src, int ldn, int nmax, int kbase) {
  int lane = threadIdx.x & 31;
  int n = lane & 15;
  if (n >= nmax) n = nmax - 1;
  int koff = (lane & 16) ? 8 : 0;
  const _Float16* p = src + (size_t)n * ldn + kbase + koff;
  v8h lo = *(const v8h*)p;          // K = koff .. koff+7
  v8h hi = *(const v8h*)(p + 16);   // K = koff+16 .. koff+23
  v16h r;
#pragma unroll
  for (int i = 0; i < 8; ++i) { r[i] = lo[i]; r[i + 8] = hi[i]; }
  return r;
}

// ---------------- elementwise f32 -> f16 ----------------
__global__ void cvt_f16_kernel(const float* __restrict__ src, _Float16* __restrict__ dst, size_t n) {
  size_t i = (size_t)blockIdx.x * 256 + threadIdx.x;
  if (i < n) dst[i] = (_Float16)src[i];
}

// zero the V token-pad region (tokens 785..799 of every (b,h,d) row)
__global__ void vpad_zero_kernel(_Float16* __restrict__ vh) {
  int i = blockIdx.x * 256 + threadIdx.x;
  const int TOT = B_ * HEADS_ * HD_ * (VSTR_ - NP_);
  if (i >= TOT) return;
  int t = i % (VSTR_ - NP_);
  int row = i / (VSTR_ - NP_);
  vh[(size_t)row * VSTR_ + NP_ + t] = (_Float16)0.0f;
}

// ---------------- fused LN1 + 2x2 attention-pool of x ----------------
__global__ __launch_bounds__(256) void lnpool_kernel(const float* __restrict__ x,
                                                     const float* __restrict__ g,
                                                     const float* __restrict__ bb,
                                                     _Float16* __restrict__ xp,
                                                     float* __restrict__ sc) {
  int p = blockIdx.x % NP_;
  int b = blockIdx.x / NP_;
  int tid = threadIdx.x;
  __shared__ float accN[C_];
  __shared__ float accR[C_];
  __shared__ float red[256];
  for (int c = tid; c < C_; c += 256) { accN[c] = 0.f; accR[c] = 0.f; }
  __syncthreads();
  int cnt = (p == 0) ? 1 : 4;
  float inv_cnt = 1.0f / (float)cnt;
  for (int s = 0; s < cnt; ++s) {
    int tok;
    if (p == 0) tok = 0;
    else {
      int pp = p - 1, y = pp / HP_, xw = pp % HP_;
      int dy = s >> 1, dx = s & 1;
      tok = 1 + (2 * y + dy) * WW_ + (2 * xw + dx);
    }
    const float* xr = x + ((size_t)b * NTOK_ + tok) * C_;
    float s1 = 0.f, s2 = 0.f;
    for (int c = tid; c < C_; c += 256) { float v = xr[c]; s1 += v; s2 += v * v; }
    red[tid] = s1; __syncthreads();
    for (int off = 128; off > 0; off >>= 1) { if (tid < off) red[tid] += red[tid + off]; __syncthreads(); }
    float mean = red[0] * (1.0f / C_); __syncthreads();
    red[tid] = s2; __syncthreads();
    for (int off = 128; off > 0; off >>= 1) { if (tid < off) red[tid] += red[tid + off]; __syncthreads(); }
    float var = red[0] * (1.0f / C_) - mean * mean; __syncthreads();
    float rstd = rsqrtf(var + 1e-6f);
    for (int c = tid; c < C_; c += 256) {
      float v = xr[c];
      accN[c] += ((v - mean) * rstd * g[c] + bb[c]) * inv_cnt;
      accR[c] += v * inv_cnt;
    }
    __syncthreads();
  }
  size_t o = ((size_t)b * NP_ + p) * C_;
  for (int c = tid; c < C_; c += 256) { xp[o + c] = (_Float16)accN[c]; sc[o + c] = accR[c]; }
}

// ---------------- generic LDS-tiled f16 WMMA GEMM, fused epilogues ----------------
// C[M,N] = A[M,K] @ B[K,N]; 64x64 block tile, 128 threads (4 waves), 16x16x32 WMMA.
// Requirements: K % 32 == 0, N % 64 == 0 (all call sites satisfy this); M-edge is
// handled by clamped staging rows + store guards.
// MODE 0: qkv scatter -> q/k f16 (b,h,t,d), v f16 transposed (b,h,d,t)
// MODE 1: + bias + skip -> d_out f32 (xo)
// MODE 2: + bias, tanh-GELU -> f16
// MODE 3: + bias, accumulate into d_out f32
template <int MODE>
__global__ __launch_bounds__(128) void gemm_kernel(const _Float16* __restrict__ A,
                                                   const _Float16* __restrict__ Bw,
                                                   int M, int Nn, int K,
                                                   float* out_f32,
                                                   _Float16* out_h0, _Float16* out_h1, _Float16* out_h2,
                                                   const float* __restrict__ bias,
                                                   const float* __restrict__ addsrc) {
  const int BK = 32;
  __shared__ __align__(16) _Float16 sA[64][BK + 8];   // row-major A tile
  __shared__ __align__(16) _Float16 sBT[64][BK + 8];  // TRANSPOSED B tile: [n][k]
  int bm = blockIdx.x * 64;
  int bn = blockIdx.y * 64;
  int tid = threadIdx.x;
  int wave = tid >> 5;
  v8f acc[4] = {v8f{}, v8f{}, v8f{}, v8f{}};

  for (int k0 = 0; k0 < K; k0 += BK) {
    // stage A: 64x32 halves = 256 x v8h; clamp row so loads are unconditional
#pragma unroll
    for (int j = tid; j < 256; j += 128) {
      int r = j >> 2, cs = (j & 3) * 8;
      int gm = bm + r; if (gm >= M) gm = M - 1;
      v8h v = *(const v8h*)(A + (size_t)gm * K + k0 + cs);
      *(v8h*)&sA[r][cs] = v;
    }
    // stage B transposed: read row-major v8h, scatter 8 halves across sBT rows
#pragma unroll
    for (int j = tid; j < 256; j += 128) {
      int k = j >> 3, ns = (j & 7) * 8;
      v8h v = *(const v8h*)(Bw + (size_t)(k0 + k) * Nn + bn + ns);
#pragma unroll
      for (int e = 0; e < 8; ++e) sBT[ns + e][k] = v[e];
    }
    if (k0 + BK < K) {  // gfx1250 global_prefetch_b8 of the next tiles
      int gm = bm + (tid >> 1); if (gm >= M) gm = M - 1;
      __builtin_prefetch(A + (size_t)gm * K + k0 + BK, 0, 1);
      __builtin_prefetch(Bw + (size_t)(k0 + BK + (tid & 31)) * Nn + bn, 0, 1);
    }
    __syncthreads();
    v16h af = frag_a(&sA[wave * 16][0], BK + 8, 16, 0);
#pragma unroll
    for (int t = 0; t < 4; ++t) {
      v16h bf = frag_b_nmajor(&sBT[t * 16][0], BK + 8, 16, 0);
      acc[t] = wmma_f16(af, bf, acc[t]);
    }
    __syncthreads();
  }

  int lane = tid & 31;
  int nl = lane & 15;
  int mo = (lane & 16) ? 8 : 0;
#pragma unroll
  for (int t = 0; t < 4; ++t) {
#pragma unroll
    for (int r = 0; r < 8; ++r) {
      int gm = bm + wave * 16 + mo + r;
      int gn = bn + t * 16 + nl;
      if (gm >= M) continue;  // store-side guard only
      float val = acc[t][r];
      if constexpr (MODE == 0) {
        int bq = gm / NP_, tk = gm % NP_;
        int s = gn / C_, hh = (gn % C_) / HD_, d = gn % HD_;
        if (s == 0)
          out_h0[(((size_t)bq * HEADS_ + hh) * NP_ + tk) * HD_ + d] = (_Float16)val;
        else if (s == 1)
          out_h1[(((size_t)bq * HEADS_ + hh) * NP_ + tk) * HD_ + d] = (_Float16)val;
        else  // V stored transposed: (b,h,d,token) with token stride VSTR_
          out_h2[(((size_t)bq * HEADS_ + hh) * HD_ + d) * VSTR_ + tk] = (_Float16)val;
      } else if constexpr (MODE == 1) {
        out_f32[(size_t)gm * C_ + gn] = val + bias[gn] + addsrc[(size_t)gm * C_ + gn];
      } else if constexpr (MODE == 2) {
        float xg = val + bias[gn];
        float gl = 0.5f * xg * (1.0f + tanhf(0.7978845608f * (xg + 0.044715f * xg * xg * xg)));
        out_h0[(size_t)gm * Nn + gn] = (_Float16)gl;
      } else {  // MODE 3
        out_f32[(size_t)gm * C_ + gn] += val + bias[gn];
      }
    }
  }
}

// ---------------- decomposed rel-pos bias tables ----------------
// rel_h[b,h,qpos,kh] = q[b,h,1+qpos,:] . rel_pos_h[qy-kh+27,:]   (qh==kh==28)
__global__ __launch_bounds__(64) void relpos_kernel(const _Float16* __restrict__ qh,
                                                    const float* __restrict__ rph,
                                                    const float* __restrict__ rpw,
                                                    float* __restrict__ relh,
                                                    float* __restrict__ relw) {
  int qpos = blockIdx.x;
  int head = blockIdx.y;
  int b = blockIdx.z;
  int tid = threadIdx.x;
  int qy = qpos / HP_, qx = qpos % HP_;
  const _Float16* qp = qh + (((size_t)(b * HEADS_ + head)) * NP_ + 1 + qpos) * HD_;
  if (tid < HP_) {
    const float* R = rph + (size_t)(qy - tid + HP_ - 1) * HD_;
    float a = 0.f;
#pragma unroll 8
    for (int d = 0; d < HD_; ++d) a += (float)qp[d] * R[d];
    relh[((size_t)(b * HEADS_ + head) * (HP_ * HP_) + qpos) * HP_ + tid] = a;
  } else if (tid >= 32 && tid < 32 + HP_) {
    int k = tid - 32;
    const float* R = rpw + (size_t)(qx - k + HP_ - 1) * HD_;
    float a = 0.f;
#pragma unroll 8
    for (int d = 0; d < HD_; ++d) a += (float)qp[d] * R[d];
    relw[((size_t)(b * HEADS_ + head) * (HP_ * HP_) + qpos) * HP_ + k] = a;
  }
}

// ---------------- fused attention: S = qK^T*s + rel, softmax, O = P V ----------------
// One workgroup per (q-tile of 16 rows, head, batch); 128 threads = 4 waves.
// Full 16x800 score row lives in LDS; P is re-materialized fragment-wise as f16
// with the softmax normalization folded into the f32->f16 conversion.
__global__ __launch_bounds__(128) void attn_kernel(const _Float16* __restrict__ qh,
                                                   const _Float16* __restrict__ kh,
                                                   const _Float16* __restrict__ vh,
                                                   const float* __restrict__ relh,
                                                   const float* __restrict__ relw,
                                                   float* __restrict__ o) {
  int qt = blockIdx.x;
  int head = blockIdx.y;
  int b = blockIdx.z;
  int q0 = qt * 16;
  int tid = threadIdx.x, wave = tid >> 5, lane = tid & 31;

  __shared__ __align__(16) float S[16][PADK_];
  __shared__ float red[16][8];
  __shared__ float rowmax[16];
  __shared__ float rowinv[16];

  const size_t base  = ((size_t)(b * HEADS_ + head)) * NP_ * HD_;   // q,k,o
  const size_t vbase = ((size_t)(b * HEADS_ + head)) * HD_ * VSTR_; // transposed V
  // Q fragments (16 rows x 64 K, two 32-K chunks); row-clamped, wide loads
  v16h a0 = frag_a(qh + base + (size_t)q0 * HD_, HD_, NP_ - q0, 0);
  v16h a1 = frag_a(qh + base + (size_t)q0 * HD_, HD_, NP_ - q0, 32);

  // Phase 1: S tiles; waves stride over the 50 k-tiles. K^T fragment is
  // contiguous in K (token-major K buffer), token index clamped.
  for (int kt = wave; kt < KT_; kt += 4) {
    int tok0 = kt * 16;
    int nlim = NP_ - tok0;
    v8f s = v8f{};
    v16h b0 = frag_b_nmajor(kh + base + (size_t)tok0 * HD_, HD_, nlim, 0);
    s = wmma_f16(a0, b0, s);
    v16h b1 = frag_b_nmajor(kh + base + (size_t)tok0 * HD_, HD_, nlim, 32);
    s = wmma_f16(a1, b1, s);
    int n = lane & 15, mo = (lane & 16) ? 8 : 0;
#pragma unroll
    for (int r = 0; r < 8; ++r) S[mo + r][tok0 + n] = s[r];
  }
  __syncthreads();

  // Phase 2: scale + rel bias + padding mask (also masks clamp artifacts)
  const float scale = 0.125f;  // 64^-0.5
  for (int i = tid; i < 16 * PADK_; i += 128) {
    int rr = i / PADK_, c = i % PADK_;
    int qi = q0 + rr;
    float v;
    if (c >= NP_ || qi >= NP_) v = -3.0e38f;
    else {
      v = S[rr][c] * scale;
      if (qi > 0 && c > 0) {
        int qp = qi - 1, kp = c - 1;
        size_t rb = ((size_t)(b * HEADS_ + head) * (HP_ * HP_) + qp) * HP_;
        v += relh[rb + kp / HP_] + relw[rb + kp % HP_];
      }
    }
    S[rr][c] = v;
  }
  __syncthreads();

  // Phase 3: softmax (8 threads per row)
  int row = tid >> 3, sub = tid & 7;
  float mx = -3.0e38f;
  for (int c = sub; c < PADK_; c += 8) mx = fmaxf(mx, S[row][c]);
  red[row][sub] = mx;
  __syncthreads();
  if (sub == 0) {
    float m2 = red[row][0];
#pragma unroll
    for (int j = 1; j < 8; ++j) m2 = fmaxf(m2, red[row][j]);
    rowmax[row] = m2;
  }
  __syncthreads();
  float rm = rowmax[row], sm = 0.f;
  for (int c = sub; c < PADK_; c += 8) { float e = __expf(S[row][c] - rm); S[row][c] = e; sm += e; }
  red[row][sub] = sm;
  __syncthreads();
  if (sub == 0) {
    float s2 = 0.f;
#pragma unroll
    for (int j = 0; j < 8; ++j) s2 += red[row][j];
    rowinv[row] = 1.0f / s2;
  }
  __syncthreads();

  // Phase 4: O = P @ V; wave w owns output columns [w*16, w*16+16).
  // V transposed (d-major) -> B fragment is contiguous wide loads; pad tokens
  // contribute P==0.
  v8f oacc = v8f{};
  for (int kc = 0; kc < PADK_ / 32; ++kc) {
    v16h ap = frag_a_f32s(&S[0][kc * 32], PADK_, rowinv);
    v16h bv = frag_b_nmajor(vh + vbase + (size_t)(wave * 16) * VSTR_ + kc * 32, VSTR_, 16, 0);
    oacc = wmma_f16(ap, bv, oacc);
  }
  int nl = lane & 15, mo = (lane & 16) ? 8 : 0;
#pragma unroll
  for (int r = 0; r < 8; ++r) {
    int gm = q0 + mo + r;
    if (gm < NP_) o[base + (size_t)gm * HD_ + wave * 16 + nl] = oacc[r];
  }
}

// ---------------- grouped 3x3 residual conv on pooled v (28x28, 12 groups of 64) ----------------
__global__ __launch_bounds__(256) void resconv_kernel(const _Float16* __restrict__ vh,
                                                      const float* __restrict__ kern,
                                                      float* __restrict__ o) {
  int idx = blockIdx.x * 256 + threadIdx.x;
  const int TOT = B_ * HP_ * HP_ * C_;
  if (idx >= TOT) return;
  int c = idx % C_;
  int p = (idx / C_) % (HP_ * HP_);
  int b = idx / (C_ * HP_ * HP_);
  int y = p / HP_, x = p % HP_;
  int hc = c / HD_, d = c % HD_;
  const _Float16* vg = vh + ((size_t)(b * HEADS_ + hc)) * HD_ * VSTR_;  // [d][token]
  float acc = 0.f;
  for (int kh = 0; kh < 3; ++kh) {
    int yy = y + kh - 1;
    if (yy < 0 || yy >= HP_) continue;
    for (int kw = 0; kw < 3; ++kw) {
      int xx = x + kw - 1;
      if (xx < 0 || xx >= HP_) continue;
      int tok = 1 + yy * HP_ + xx;
      const float* kp = kern + ((size_t)(kh * 3 + kw) * HD_) * C_ + c;
#pragma unroll 8
      for (int i = 0; i < HD_; ++i) acc += (float)vg[(size_t)i * VSTR_ + tok] * kp[(size_t)i * C_];
    }
  }
  o[(((size_t)(b * HEADS_ + hc)) * NP_ + 1 + p) * HD_ + d] += acc;
}

// ---------------- pack o (b,h,t,d) f32 -> token-major f16, add cls residual ----------------
__global__ __launch_bounds__(256) void pack_o_kernel(const float* __restrict__ o,
                                                     const _Float16* __restrict__ vh,
                                                     _Float16* __restrict__ oT) {
  int idx = blockIdx.x * 256 + threadIdx.x;
  const int TOT = MTOK_ * C_;
  if (idx >= TOT) return;
  int c = idx % C_;
  int m = idx / C_;
  int b = m / NP_, t = m % NP_;
  int hc = c / HD_, d = c % HD_;
  float val = o[(((size_t)(b * HEADS_ + hc)) * NP_ + t) * HD_ + d];
  if (t == 0) val += (float)vh[(((size_t)(b * HEADS_ + hc)) * HD_ + d) * VSTR_];
  oT[idx] = (_Float16)val;
}

// ---------------- LN2 over xo (read from d_out) -> f16 ----------------
__global__ __launch_bounds__(256) void ln2_kernel(const float* __restrict__ xo,
                                                  const float* __restrict__ g,
                                                  const float* __restrict__ bb,
                                                  _Float16* __restrict__ out) {
  int m = blockIdx.x;
  int tid = threadIdx.x;
  __shared__ float red[256];
  const float* xr = xo + (size_t)m * C_;
  float s1 = 0.f, s2 = 0.f;
  for (int c = tid; c < C_; c += 256) { float v = xr[c]; s1 += v; s2 += v * v; }
  red[tid] = s1; __syncthreads();
  for (int off = 128; off > 0; off >>= 1) { if (tid < off) red[tid] += red[tid + off]; __syncthreads(); }
  float mean = red[0] * (1.0f / C_); __syncthreads();
  red[tid] = s2; __syncthreads();
  for (int off = 128; off > 0; off >>= 1) { if (tid < off) red[tid] += red[tid + off]; __syncthreads(); }
  float var = red[0] * (1.0f / C_) - mean * mean; __syncthreads();
  float rstd = rsqrtf(var + 1e-6f);
  for (int c = tid; c < C_; c += 256)
    out[(size_t)m * C_ + c] = (_Float16)((xr[c] - mean) * rstd * g[c] + bb[c]);
}

// ---------------- launcher ----------------
extern "C" void kernel_launch(void* const* d_in, const int* in_sizes, int n_in,
                              void* d_out, int out_size, void* d_ws, size_t ws_size,
                              hipStream_t stream) {
  const float* x        = (const float*)d_in[0];
  const float* W_qkv    = (const float*)d_in[1];
  const float* W_proj   = (const float*)d_in[2];
  const float* b_proj   = (const float*)d_in[3];
  const float* rel_pos_h= (const float*)d_in[4];
  const float* rel_pos_w= (const float*)d_in[5];
  const float* res_kern = (const float*)d_in[6];
  const float* ln1_g    = (const float*)d_in[7];
  const float* ln1_b    = (const float*)d_in[8];
  const float* ln2_g    = (const float*)d_in[9];
  const float* ln2_b    = (const float*)d_in[10];
  const float* W1       = (const float*)d_in[11];
  const float* b1       = (const float*)d_in[12];
  const float* W2       = (const float*)d_in[13];
  const float* b2       = (const float*)d_in[14];
  float* out = (float*)d_out;

  // workspace carve-up (256-B aligned blocks)
  char* w = (char*)d_ws;
  auto alloc = [&](size_t bytes) -> void* {
    void* p = (void*)w;
    w += (bytes + 255) & ~(size_t)255;
    return p;
  };
  const size_t NE = (size_t)MTOK_ * C_;               // 4,823,040
  _Float16* xp_h   = (_Float16*)alloc(NE * 2);
  float*    sc     = (float*)   alloc(NE * 4);
  _Float16* wqkv_h = (_Float16*)alloc((size_t)C_ * 3 * C_ * 2);
  _Float16* wproj_h= (_Float16*)alloc((size_t)C_ * C_ * 2);
  _Float16* w1_h   = (_Float16*)alloc((size_t)C_ * 4 * C_ * 2);
  _Float16* w2_h   = (_Float16*)alloc((size_t)4 * C_ * C_ * 2);
  _Float16* q_h    = (_Float16*)alloc(NE * 2);
  _Float16* k_h    = (_Float16*)alloc(NE * 2);
  _Float16* v_h    = (_Float16*)alloc((size_t)B_ * HEADS_ * HD_ * VSTR_ * 2);  // transposed+padded
  float*    relh   = (float*)   alloc((size_t)B_ * HEADS_ * HP_ * HP_ * HP_ * 4);
  float*    relw   = (float*)   alloc((size_t)B_ * HEADS_ * HP_ * HP_ * HP_ * 4);
  float*    o_buf  = (float*)   alloc(NE * 4);
  _Float16* oT_h   = (_Float16*)alloc(NE * 2);
  _Float16* ln2_h  = (_Float16*)alloc(NE * 2);
  _Float16* h1_h   = (_Float16*)alloc((size_t)MTOK_ * 4 * C_ * 2);
  (void)in_sizes; (void)n_in; (void)out_size; (void)ws_size;

  // 1) weights -> f16, V pad -> 0
  {
    size_t n = (size_t)C_ * 3 * C_;
    cvt_f16_kernel<<<(n + 255) / 256, 256, 0, stream>>>(W_qkv, wqkv_h, n);
    n = (size_t)C_ * C_;
    cvt_f16_kernel<<<(n + 255) / 256, 256, 0, stream>>>(W_proj, wproj_h, n);
    n = (size_t)C_ * 4 * C_;
    cvt_f16_kernel<<<(n + 255) / 256, 256, 0, stream>>>(W1, w1_h, n);
    cvt_f16_kernel<<<(n + 255) / 256, 256, 0, stream>>>(W2, w2_h, n);
    int tot = B_ * HEADS_ * HD_ * (VSTR_ - NP_);
    vpad_zero_kernel<<<(tot + 255) / 256, 256, 0, stream>>>(v_h);
  }
  // 2) fused LN1 + pool (pooling commutes with the qkv projection)
  lnpool_kernel<<<B_ * NP_, 256, 0, stream>>>(x, ln1_g, ln1_b, xp_h, sc);
  // 3) qkv GEMM 6280x2304x768, scatter to q/k (b,h,t,d) and v (b,h,d,t)
  {
    dim3 grid((MTOK_ + 63) / 64, (3 * C_) / 64);
    gemm_kernel<0><<<grid, 128, 0, stream>>>(xp_h, wqkv_h, MTOK_, 3 * C_, C_,
                                             nullptr, q_h, k_h, v_h, nullptr, nullptr);
  }
  // 4) rel-pos tables
  relpos_kernel<<<dim3(HP_ * HP_, HEADS_, B_), 64, 0, stream>>>(q_h, rel_pos_h, rel_pos_w, relh, relw);
  // 5) fused attention (per 16 q-rows x head x batch)
  attn_kernel<<<dim3(KT_, HEADS_, B_), 128, 0, stream>>>(q_h, k_h, v_h, relh, relw, o_buf);
  // 6) grouped residual conv accumulated into o
  {
    int tot = B_ * HP_ * HP_ * C_;
    resconv_kernel<<<(tot + 255) / 256, 256, 0, stream>>>(v_h, res_kern, o_buf);
  }
  // 7) pack o -> token-major f16 (+cls residual)
  pack_o_kernel<<<((int)NE + 255) / 256, 256, 0, stream>>>(o_buf, v_h, oT_h);
  // 8) projection GEMM, + b_proj + pooled skip -> d_out (xo)
  {
    dim3 grid((MTOK_ + 63) / 64, C_ / 64);
    gemm_kernel<1><<<grid, 128, 0, stream>>>(oT_h, wproj_h, MTOK_, C_, C_,
                                             out, nullptr, nullptr, nullptr, b_proj, sc);
  }
  // 9) LN2 on xo
  ln2_kernel<<<MTOK_, 256, 0, stream>>>(out, ln2_g, ln2_b, ln2_h);
  // 10) MLP up GEMM + GELU
  {
    dim3 grid((MTOK_ + 63) / 64, (4 * C_) / 64);
    gemm_kernel<2><<<grid, 128, 0, stream>>>(ln2_h, w1_h, MTOK_, 4 * C_, C_,
                                             nullptr, h1_h, nullptr, nullptr, b1, nullptr);
  }
  // 11) MLP down GEMM, accumulate into d_out
  {
    dim3 grid((MTOK_ + 63) / 64, C_ / 64);
    gemm_kernel<3><<<grid, 128, 0, stream>>>(h1_h, w2_h, MTOK_, C_, 4 * C_,
                                             out, nullptr, nullptr, nullptr, b2, nullptr);
  }
}
